// TransformerLayer_30760555774503
// MI455X (gfx1250) — matrix operand order repaired and checked by reading, not compiled, not run on hardware
//
#include <hip/hip_runtime.h>

// ---------------------------------------------------------------------------
// MI455X (gfx1250, wave32) transformer layer:
//   LN1 -> QKV gemm -> block-causal windowed attention -> Wo gemm (+resid)
//   -> LN2 -> FFN1 (+GELU) -> FFN2 (+resid) -> d_out (f32)
// Matrix math: V_WMMA_F32_16X16X32_BF16 (f32 accumulate).
// Data movement: GLOBAL_LOAD_ASYNC_TO_LDS_B128 (ASYNCcnt), double-buffered
// and software-pipelined with static buffer indices (K % 64 == 0).
// ---------------------------------------------------------------------------

typedef __bf16 bf16;
typedef __attribute__((ext_vector_type(16))) __bf16 v16bf;
typedef __attribute__((ext_vector_type(8)))  float  v8f;

#define B_  2
#define S_  4096
#define D_  768
#define H_  12
#define HD_ 64
#define WS_ 256
#define NW_ 16

__device__ __forceinline__ v8f wmma_bf16(v16bf a, v16bf b, v8f c) {
  return __builtin_amdgcn_wmma_f32_16x16x32_bf16(false, a, false, b, (short)0, c,
                                                 false, false);
}

// Low 32 bits of a flat pointer into LDS == byte offset from the wave's
// LDS base (shared aperture lives entirely in addr[63:32] on gfx1250).
__device__ __forceinline__ unsigned lds_off(const void* p) {
  return (unsigned)(size_t)p;
}

// Async global->LDS copies. INST_OFFSET is added to BOTH the global and the
// LDS address, so one (lds, global) pair covers a contiguous per-lane run.
__device__ __forceinline__ void async_b128_x2(unsigned lds, const void* g) {
  asm volatile(
      "global_load_async_to_lds_b128 %0, %1, off\n\t"
      "global_load_async_to_lds_b128 %0, %1, off offset:16"
      :: "v"(lds), "v"(g) : "memory");
}
__device__ __forceinline__ void async_b128_x4(unsigned lds, const void* g) {
  asm volatile(
      "global_load_async_to_lds_b128 %0, %1, off\n\t"
      "global_load_async_to_lds_b128 %0, %1, off offset:16\n\t"
      "global_load_async_to_lds_b128 %0, %1, off offset:32\n\t"
      "global_load_async_to_lds_b128 %0, %1, off offset:48"
      :: "v"(lds), "v"(g) : "memory");
}
__device__ __forceinline__ void async_b128_x8(unsigned lds, const void* g) {
  asm volatile(
      "global_load_async_to_lds_b128 %0, %1, off\n\t"
      "global_load_async_to_lds_b128 %0, %1, off offset:16\n\t"
      "global_load_async_to_lds_b128 %0, %1, off offset:32\n\t"
      "global_load_async_to_lds_b128 %0, %1, off offset:48\n\t"
      "global_load_async_to_lds_b128 %0, %1, off offset:64\n\t"
      "global_load_async_to_lds_b128 %0, %1, off offset:80\n\t"
      "global_load_async_to_lds_b128 %0, %1, off offset:96\n\t"
      "global_load_async_to_lds_b128 %0, %1, off offset:112"
      :: "v"(lds), "v"(g) : "memory");
}
__device__ __forceinline__ void wait_async0() {
  asm volatile("s_wait_asynccnt 0" ::: "memory");
}

// A fragment: 16x32 bf16, row-major, leading dim `ld`. Layout: lane
// m=l&15, h=l>>4; VGPR v0..7 -> K=2v+16h(+1) (matches frag_b).
__device__ __forceinline__ v16bf frag_a(const bf16* src, int ld, int lane) {
  const int m = lane & 15, h = lane >> 4;
  const bf16* r = src + m * ld;
  union { unsigned u[8]; v16bf v; } f;
#pragma unroll
  for (int i = 0; i < 4; ++i) f.u[i]     = *(const unsigned*)(r + 2 * i + 16 * h);
#pragma unroll
  for (int i = 0; i < 4; ++i) f.u[4 + i] = *(const unsigned*)(r + 8 + 2 * i + 16 * h);
  return f.v;
}

// B fragment: 32x16 (KxN) read from TRANSPOSED storage srcT[n*ld + k].
// Layout (as frag_a): lane n=l&15, K = 2v+16h(+1) -> contiguous b32 pairs.
__device__ __forceinline__ v16bf frag_b(const bf16* srcT, int ld, int lane) {
  const int n = lane & 15, h = lane >> 4;
  const bf16* r = srcT + n * ld + 16 * h;
  union { unsigned u[8]; v16bf v; } f;
#pragma unroll
  for (int i = 0; i < 8; ++i) f.u[i] = *(const unsigned*)(r + 2 * i);
  return f.v;
}

// ---------------------------------------------------------------------------
// bf16 WMMA GEMM, B supplied pre-transposed ([N][K] row-major) so BOTH tiles
// are contiguous async copies. 128x128x32 tiles, double-buffered LDS with a
// 2x-unrolled software pipeline (static buffer indices), 8 waves (4x2),
// 8 16x16 accum tiles/wave, one barrier per K-step. Requires K % 64 == 0.
// ---------------------------------------------------------------------------
__global__ __launch_bounds__(256) void gemm_bf16_wmma(
    const bf16* __restrict__ A, const bf16* __restrict__ BT,
    const float* __restrict__ bias, const float* __restrict__ resid,
    float* __restrict__ Cf, bf16* __restrict__ Cb,
    int M, int N, int K, int do_gelu)
{
  (void)M;
  constexpr int BM = 128, BN = 128, BK = 32;
  __shared__ __align__(16) bf16 As[2][BM][BK];   // 2 x 8 KB
  __shared__ __align__(16) bf16 Bs[2][BN][BK];   // 2 x 8 KB ([n][k])
  const int t = threadIdx.x, lane = t & 31, w = t >> 5;
  const int wr = w >> 1, wc = w & 1;              // waves: 4 rows x 2 cols
  const int bn0 = blockIdx.x * BN, bm0 = blockIdx.y * BM;

  // each thread owns one contiguous 32B run per tile (row, 16-elem segment)
  const int crow = t >> 1, cseg = (t & 1) * 16;
  const bf16* gA = A  + (size_t)(bm0 + crow) * K + cseg;
  const bf16* gB = BT + (size_t)(bn0 + crow) * K + cseg;
  const unsigned lA0 = lds_off(&As[0][crow][cseg]), lA1 = lds_off(&As[1][crow][cseg]);
  const unsigned lB0 = lds_off(&Bs[0][crow][cseg]), lB1 = lds_off(&Bs[1][crow][cseg]);

  v8f z = {0.f,0.f,0.f,0.f,0.f,0.f,0.f,0.f};
  v8f acc[2][4];
#pragma unroll
  for (int i = 0; i < 2; ++i)
#pragma unroll
    for (int j = 0; j < 4; ++j) acc[i][j] = z;

  // one pipeline phase: consume buffer `buf`, issue async stage of tile knext
  auto phase = [&](int buf, int knext, bool do_issue) {
    wait_async0();        // own async writes done ...
    __syncthreads();      // ... then everyone's tile visible / prev reads done
    if (do_issue) {
      async_b128_x2(buf ? lA0 : lA1, gA + knext);
      async_b128_x2(buf ? lB0 : lB1, gB + knext);
    }
    v16bf af[2], bfv[4];
#pragma unroll
    for (int i = 0; i < 2; ++i) af[i]  = frag_a(&As[buf][wr * 32 + i * 16][0], BK, lane);
#pragma unroll
    for (int j = 0; j < 4; ++j) bfv[j] = frag_b(&Bs[buf][wc * 64 + j * 16][0], BK, lane);
#pragma unroll
    for (int i = 0; i < 2; ++i)
#pragma unroll
      for (int j = 0; j < 4; ++j)
        acc[i][j] = wmma_bf16(af[i], bfv[j], acc[i][j]);
  };

  // prologue: stage tile 0 into buf0
  async_b128_x2(lA0, gA);
  async_b128_x2(lB0, gB);

  // K % 64 == 0: phase0 always has a next tile; phase1's guard is the only one
  for (int kk = 0; kk < K; kk += 2 * BK) {
    phase(0, kk + BK, true);                 // consume buf0, stage buf1
    phase(1, kk + 2 * BK, kk + 2 * BK < K);  // consume buf1, stage buf0
  }

  // epilogue; C layout: lane col = l&15, rows r + 8*(l>>4) across VGPRs
  const int ln = lane & 15, hh = lane >> 4;
#pragma unroll
  for (int i = 0; i < 2; ++i) {
#pragma unroll
    for (int j = 0; j < 4; ++j) {
      const int col = bn0 + wc * 64 + j * 16 + ln;
      const float bia = bias ? bias[col] : 0.f;
#pragma unroll
      for (int r = 0; r < 8; ++r) {
        const int row = bm0 + wr * 32 + i * 16 + r + 8 * hh;
        float v = acc[i][j][r] + bia;
        if (do_gelu) v = 0.5f * v * (1.f + erff(v * 0.70710678118f)); // exact GELU
        if (resid)   v += resid[(size_t)row * N + col];
        if (Cf) Cf[(size_t)row * N + col] = v;
        if (Cb) Cb[(size_t)row * N + col] = (bf16)v;
      }
    }
  }
}

// ---------------------------------------------------------------------------
// Block-causal windowed attention. Workgroup = (b, head, half-window of 128 q
// rows), 8 waves x 16 rows. Q/K staged via async-to-LDS; scores in registers
// (16 col-tiles); per-block softmax via shfl_xor; V^T restaged over K's LDS;
// ctx += P@V via WMMA (P through per-wave LDS). LDS: 16K+32K+8K = 56 KB.
// ---------------------------------------------------------------------------
__global__ __launch_bounds__(256) void attn_win_wmma(
    const bf16* __restrict__ qkv, bf16* __restrict__ ctxb)
{
  __shared__ __align__(16) bf16 Qs[128][64];
  __shared__ __align__(16) bf16 KV[256 * 64];   // K as [key][64], then V^T [hd][256]
  __shared__ __align__(16) bf16 Pw[8][16][32];  // per-wave P staging

  const int t = threadIdx.x, lane = t & 31, w = t >> 5;
  const int qw = blockIdx.x >> 1, hsel = blockIdx.x & 1;
  const int head = blockIdx.y, b = blockIdx.z;
  const size_t ldq = 3 * D_;                    // 2304
  const int qrow0 = w * 16;

  {   // Q half-window -> LDS (async, 64B per thread)
    const int row = t >> 1, seg = (t & 1) * 32;
    const bf16* src = qkv + ((size_t)b * S_ + (size_t)qw * WS_ + hsel * 128 + row) * ldq
                          + head * HD_ + seg;
    async_b128_x4(lds_off(&Qs[row][seg]), src);
  }
  wait_async0();
  __syncthreads();

  v8f z = {0.f,0.f,0.f,0.f,0.f,0.f,0.f,0.f};
  v8f ctx[4];
#pragma unroll
  for (int i = 0; i < 4; ++i) ctx[i] = z;
  const float inv = 0.125f;                     // 1/sqrt(HD)

  for (int j = 0; j <= qw; ++j) {
    __syncthreads();   // prior iteration's V^T reads complete before overwrite
    {   // K_j -> KV[key][64] (async, 128B per thread)
      const bf16* src = qkv + ((size_t)b * S_ + (size_t)j * WS_ + t) * ldq
                            + D_ + head * HD_;
      async_b128_x8(lds_off(&KV[t * 64]), src);
    }
    wait_async0();
    __syncthreads();

    // scores: S = Q K^T ; KV[key][hd] is already transposed-B storage
    v16bf qa0 = frag_a(&Qs[qrow0][0],  64, lane);
    v16bf qa1 = frag_a(&Qs[qrow0][32], 64, lane);
    v8f st[16];
#pragma unroll
    for (int ct = 0; ct < 16; ++ct) {
      v16bf kb0 = frag_b(&KV[ct * 16 * 64],      64, lane);
      v16bf kb1 = frag_b(&KV[ct * 16 * 64 + 32], 64, lane);
      v8f s = z;
      s = wmma_bf16(qa0, kb0, s);
      s = wmma_bf16(qa1, kb1, s);
      st[ct] = s;
    }
    __syncthreads();   // everyone done reading K before V^T overwrites

    {   // V_j^T -> KV[hd][key] (needs transpose -> scalar scatter)
      const bf16* src = qkv + ((size_t)b * S_ + (size_t)j * WS_ + t) * ldq
                            + 2 * D_ + head * HD_;
      const uint4* g = (const uint4*)src;
#pragma unroll
      for (int c8 = 0; c8 < 8; ++c8) {
        uint4 d = g[c8];
        const bf16* e = (const bf16*)&d;
#pragma unroll
        for (int i = 0; i < 8; ++i) KV[(c8 * 8 + i) * 256 + t] = e[i];
      }
    }

    // per-row block softmax: row r lives in VGPR slot r across a 16-lane half
    float mrow[8], srow[8], rs[8];
#pragma unroll
    for (int r = 0; r < 8; ++r) {
      float m = st[0][r];
#pragma unroll
      for (int ct = 1; ct < 16; ++ct) m = fmaxf(m, st[ct][r]);
#pragma unroll
      for (int off = 1; off <= 8; off <<= 1) m = fmaxf(m, __shfl_xor(m, off, 32));
      mrow[r] = m;
      srow[r] = 0.f;
    }
#pragma unroll
    for (int ct = 0; ct < 16; ++ct)
#pragma unroll
      for (int r = 0; r < 8; ++r) {
        float pv = __expf((st[ct][r] - mrow[r]) * inv);
        st[ct][r] = pv;
        srow[r] += pv;
      }
#pragma unroll
    for (int r = 0; r < 8; ++r) {
      float s = srow[r];
#pragma unroll
      for (int off = 1; off <= 8; off <<= 1) s += __shfl_xor(s, off, 32);
      rs[r] = 1.f / s;
    }
    __syncthreads();   // V^T resident for all waves

    // ctx += P @ V over 8 chunks of 32 keys; P through wave-private LDS
    for (int kc = 0; kc < 8; ++kc) {
#pragma unroll
      for (int tt = 0; tt < 2; ++tt) {
        const int ct = kc * 2 + tt;
#pragma unroll
        for (int r = 0; r < 8; ++r)
          Pw[w][r + 8 * (lane >> 4)][tt * 16 + (lane & 15)] =
              (bf16)(st[ct][r] * rs[r]);
      }
      v16bf pa = frag_a(&Pw[w][0][0], 32, lane);
#pragma unroll
      for (int ht = 0; ht < 4; ++ht) {
        v16bf vb = frag_b(&KV[ht * 16 * 256 + kc * 32], 256, lane);
        ctx[ht] = wmma_bf16(pa, vb, ctx[ht]);
      }
    }
  }

  // store ctx -> [B*S, 768] bf16 (feeds Wo GEMM)
  const int qglob = qw * WS_ + hsel * 128 + qrow0;
#pragma unroll
  for (int ht = 0; ht < 4; ++ht) {
    const int col = head * HD_ + ht * 16 + (lane & 15);
#pragma unroll
    for (int r = 0; r < 8; ++r) {
      const size_t row = (size_t)b * S_ + qglob + r + 8 * (lane >> 4);
      ctxb[row * D_ + col] = (bf16)ctx[ht][r];
    }
  }
}

// ---------------------------------------------------------------------------
// LayerNorm: one wave per row of 768, shuffle reductions, bf16 out.
// ---------------------------------------------------------------------------
__global__ __launch_bounds__(256) void layernorm_to_bf16(
    const float* __restrict__ x, const float* __restrict__ g,
    const float* __restrict__ bv, bf16* __restrict__ out)
{
  const int lane = threadIdx.x & 31, w = threadIdx.x >> 5;
  const size_t row = (size_t)blockIdx.x * 8 + w;
  const float* xr = x + row * D_;
  float vals[24];
  float s = 0.f;
#pragma unroll
  for (int i = 0; i < 24; ++i) { vals[i] = xr[lane + 32 * i]; s += vals[i]; }
#pragma unroll
  for (int off = 1; off < 32; off <<= 1) s += __shfl_xor(s, off, 32);
  const float mu = s * (1.f / 768.f);
  float v2 = 0.f;
#pragma unroll
  for (int i = 0; i < 24; ++i) { float d = vals[i] - mu; v2 += d * d; }
#pragma unroll
  for (int off = 1; off < 32; off <<= 1) v2 += __shfl_xor(v2, off, 32);
  const float rstd = rsqrtf(v2 * (1.f / 768.f) + 1e-5f);
  bf16* orow = out + row * D_;
#pragma unroll
  for (int i = 0; i < 24; ++i) {
    const int c = lane + 32 * i;
    orow[c] = (bf16)((vals[i] - mu) * rstd * g[c] + bv[c]);
  }
}

// -------------------- weight conversion (bf16, TRANSPOSED) -----------------
// out[N][K] = in[K][N]^T : makes GEMM B-tiles contiguous (async-copy friendly)
__global__ void cvt_f32_to_bf16_T(const float* __restrict__ in, bf16* __restrict__ out,
                                  int N, int K) {
  int i = blockIdx.x * 256 + threadIdx.x;
  if (i >= N * K) return;
  int n = i / K, k = i % K;
  out[i] = (bf16)in[(size_t)k * N + n];
}
__global__ void pack_qkv_wT(const float* __restrict__ wq, const float* __restrict__ wk,
                            const float* __restrict__ wv, bf16* __restrict__ out) {
  int i = blockIdx.x * 256 + threadIdx.x;
  if (i >= 2304 * 768) return;
  int c = i / 768, k = i % 768;   // out[c][k] = W[k][c]
  float v = (c < 768) ? wq[(size_t)k * 768 + c]
          : (c < 1536) ? wk[(size_t)k * 768 + (c - 768)]
                       : wv[(size_t)k * 768 + (c - 1536)];
  out[i] = (bf16)v;
}
__global__ void pack_qkv_b(const float* __restrict__ bq, const float* __restrict__ bk,
                           const float* __restrict__ bvv, float* __restrict__ out) {
  int i = blockIdx.x * 256 + threadIdx.x;
  if (i >= 2304) return;
  out[i] = (i < 768) ? bq[i] : (i < 1536) ? bk[i - 768] : bvv[i - 1536];
}

// ---------------------------------------------------------------------------
extern "C" void kernel_launch(void* const* d_in, const int* in_sizes, int n_in,
                              void* d_out, int out_size, void* d_ws, size_t ws_size,
                              hipStream_t stream) {
  (void)in_sizes; (void)n_in; (void)out_size; (void)ws_size;
  const float* x    = (const float*)d_in[0];
  const float* Wq   = (const float*)d_in[1];
  const float* bq   = (const float*)d_in[2];
  const float* Wk   = (const float*)d_in[3];
  const float* bk   = (const float*)d_in[4];
  const float* Wv   = (const float*)d_in[5];
  const float* bv   = (const float*)d_in[6];
  const float* Wo   = (const float*)d_in[7];
  const float* bo   = (const float*)d_in[8];
  const float* ln1g = (const float*)d_in[9];
  const float* ln1b = (const float*)d_in[10];
  const float* W1   = (const float*)d_in[11];
  const float* b1   = (const float*)d_in[12];
  const float* W2   = (const float*)d_in[13];
  const float* b2   = (const float*)d_in[14];
  const float* ln2g = (const float*)d_in[15];
  const float* ln2b = (const float*)d_in[16];

  const size_t MR = (size_t)B_ * S_;   // 8192 rows
  char* p = (char*)d_ws;
  size_t off = 0;
  auto take = [&](size_t bytes) -> char* {
    char* r = p + off;
    off = (off + bytes + 255) & ~(size_t)255;
    return r;
  };
  bf16*  WqkvT = (bf16*)take((size_t)2304 * 768 * 2);
  float* bqkv  = (float*)take(2304 * 4);
  bf16*  WoT   = (bf16*)take((size_t)768 * 768 * 2);
  bf16*  W1T   = (bf16*)take((size_t)3072 * 768 * 2);
  bf16*  W2T   = (bf16*)take((size_t)768 * 3072 * 2);
  bf16*  hbuf  = (bf16*)take(MR * 768 * 2);    // LN output (reused for LN2)
  bf16*  ctxb  = (bf16*)take(MR * 768 * 2);    // attention context
  float* x1    = (float*)take(MR * 768 * 4);   // x + attn_out (f32 residual)
  char*  big   = take(MR * 3072 * 2);          // union: qkv / gelu activations
  bf16*  qkvb  = (bf16*)big;                   // dead before gbuf is written
  bf16*  gbuf  = (bf16*)big;

  // --- weight prep (bf16, transposed for contiguous GEMM B tiles) ---
  { int n = 2304 * 768; pack_qkv_wT<<<(n + 255) / 256, 256, 0, stream>>>(Wq, Wk, Wv, WqkvT); }
  pack_qkv_b<<<9, 256, 0, stream>>>(bq, bk, bv, bqkv);
  { int n = 768 * 768;  cvt_f32_to_bf16_T<<<(n + 255) / 256, 256, 0, stream>>>(Wo, WoT, 768, 768); }
  { int n = 3072 * 768; cvt_f32_to_bf16_T<<<(n + 255) / 256, 256, 0, stream>>>(W1, W1T, 3072, 768); }
  { int n = 768 * 3072; cvt_f32_to_bf16_T<<<(n + 255) / 256, 256, 0, stream>>>(W2, W2T, 768, 3072); }

  // --- LN1 ---
  layernorm_to_bf16<<<MR / 8, 256, 0, stream>>>(x, ln1g, ln1b, hbuf);

  // --- fused QKV projection: [8192,768] x [768,2304] ---
  gemm_bf16_wmma<<<dim3(2304 / 128, MR / 128), 256, 0, stream>>>(
      hbuf, WqkvT, bqkv, nullptr, nullptr, qkvb, (int)MR, 2304, 768, 0);

  // --- windowed block-causal attention ---
  attn_win_wmma<<<dim3(NW_ * 2, H_, B_), 256, 0, stream>>>(qkvb, ctxb);

  // --- output projection + residual: x1 = x + ctx @ Wo + bo ---
  gemm_bf16_wmma<<<dim3(768 / 128, MR / 128), 256, 0, stream>>>(
      ctxb, WoT, bo, x, x1, nullptr, (int)MR, 768, 768, 0);

  // --- LN2 ---
  layernorm_to_bf16<<<MR / 8, 256, 0, stream>>>(x1, ln2g, ln2b, hbuf);

  // --- FFN1 + exact GELU: [8192,768] x [768,3072] ---
  gemm_bf16_wmma<<<dim3(3072 / 128, MR / 128), 256, 0, stream>>>(
      hbuf, W1T, b1, nullptr, nullptr, gbuf, (int)MR, 3072, 768, 1);

  // --- FFN2 + residual -> d_out (f32): [8192,3072] x [3072,768] ---
  gemm_bf16_wmma<<<dim3(768 / 128, MR / 128), 256, 0, stream>>>(
      gbuf, W2T, b2, x1, (float*)d_out, nullptr, (int)MR, 768, 3072, 0);
}
